// NCA_57801669869630
// MI455X (gfx1250) — matrix operand order repaired and checked by reading, not compiled, not run on hardware
//
#include <hip/hip_runtime.h>

typedef __attribute__((ext_vector_type(16))) _Float16 v16h;
typedef __attribute__((ext_vector_type(8)))  float    v8f;

#define B_   8
#define C_   16
#define H_   128
#define W_   128
#define HID  128
#define CLAMP_V  5.0f
#define ALIVE_TH 0.1f

#define AS1 __attribute__((address_space(1)))
#define AS3 __attribute__((address_space(3)))

#if defined(__has_builtin)
#  if __has_builtin(__builtin_amdgcn_global_load_async_to_lds_b32)
#    define USE_ASYNC_LDS 1
#  endif
#endif
#ifndef USE_ASYNC_LDS
#  define USE_ASYNC_LDS 0
#endif

#if USE_ASYNC_LDS
__device__ __forceinline__ void async_g2l_b32(const float* g, float* l) {
  __builtin_amdgcn_global_load_async_to_lds_b32(
      (AS1 int*)(void*)g, (AS3 int*)(void*)l, 0, 0);
}
__device__ __forceinline__ void wait_async0() {
#if __has_builtin(__builtin_amdgcn_s_wait_asynccnt)
  __builtin_amdgcn_s_wait_asynccnt(0);
#else
  asm volatile("s_wait_asynccnt 0x0" ::: "memory");
#endif
}
#endif

// ---------------------------------------------------------------------------
// Prep: convert w1 (128x48, padded K->64) and w2 (16x128) into f16 WMMA
// A-fragment order: frag f = ((mt*2+kh) or j), element ((f*32+lane)*16+half).
// A 16x32 f16 layout: lane<16: halves0..7=K0..7, 8..15=K16..23;
//                     lane>=16: halves0..7=K8..15, 8..15=K24..31; M=lane&15.
// ---------------------------------------------------------------------------
__global__ void nca_prep(const float* __restrict__ w1, const float* __restrict__ w2,
                         _Float16* __restrict__ w1A, _Float16* __restrict__ w2A) {
  const int tid = threadIdx.x;
  for (int i = tid; i < 8 * 2 * 32 * 16; i += 256) {
    int half = i & 15;
    int lane = (i >> 4) & 31;
    int frag = i >> 9;                // mt*2 + kh
    int kh = frag & 1, mt = frag >> 1;
    int hid = 16 * mt + (lane & 15);
    int kl = (half < 8 ? half : half + 8) + ((lane >> 4) ? 8 : 0);
    int k = 32 * kh + kl;
    w1A[i] = (k < 48) ? (_Float16)w1[hid * 48 + k] : (_Float16)0.0f;
  }
  for (int i = tid; i < 4 * 32 * 16; i += 256) {
    int half = i & 15;
    int lane = (i >> 4) & 31;
    int j = i >> 9;
    int o = lane & 15;
    int kl = (half < 8 ? half : half + 8) + ((lane >> 4) ? 8 : 0);
    int k = 32 * j + kl;
    w2A[i] = (_Float16)w2[o * 128 + k];
  }
}

// ---------------------------------------------------------------------------
// Step kernel: perceive (sobel) + 2-layer pointwise MLP via WMMA + residual.
// Block = 128 threads = 4 waves; wave handles one row of 16 pixels.
// Halo tile + biases staged via GLOBAL_LOAD_ASYNC_TO_LDS when available.
// ---------------------------------------------------------------------------
__global__ __launch_bounds__(128)
void nca_step(const float* __restrict__ s_in, float* __restrict__ s_mid,
              unsigned char* __restrict__ pre_mask,
              const _Float16* __restrict__ w1A, const _Float16* __restrict__ w2A,
              const float* __restrict__ b1, const float* __restrict__ b2) {
  __shared__ float sl[C_][6][18];                        // state tile + halo
  __shared__ float bl1[HID];
  __shared__ float bl2[C_];
  __shared__ __align__(32) _Float16 hl[4][HID][16];      // per-wave h staging

  const int tid  = threadIdx.x;
  const int wv   = tid >> 5;
  const int lane = tid & 31;
  const int n    = lane & 15;      // pixel within wave tile (N index)
  const int hi   = lane >> 4;      // half-wave select
  const int x0 = blockIdx.x * 16, y0 = blockIdx.y * 4, b = blockIdx.z;
  const int y = y0 + wv;

  // ---- stage state halo tile + biases into LDS ----
#if USE_ASYNC_LDS
  async_g2l_b32(&b1[tid], &bl1[tid]);
  if (tid < C_) async_g2l_b32(&b2[tid], &bl2[tid]);
  for (int i = tid; i < C_ * 6 * 18; i += 128) {
    int ch = i / 108, rem = i % 108, r = rem / 18, cc = rem % 18;
    int gy = y0 - 1 + r, gx = x0 - 1 + cc;
    if (gy >= 0 && gy < H_ && gx >= 0 && gx < W_)
      async_g2l_b32(&s_in[((b * C_ + ch) * H_ + gy) * W_ + gx], &sl[ch][r][cc]);
    else
      sl[ch][r][cc] = 0.0f;
  }
  wait_async0();
#else
  bl1[tid] = b1[tid];
  if (tid < C_) bl2[tid] = b2[tid];
  for (int i = tid; i < C_ * 6 * 18; i += 128) {
    int ch = i / 108, rem = i % 108, r = rem / 18, cc = rem % 18;
    int gy = y0 - 1 + r, gx = x0 - 1 + cc;
    float v = 0.0f;
    if (gy >= 0 && gy < H_ && gx >= 0 && gx < W_)
      v = s_in[((b * C_ + ch) * H_ + gy) * W_ + gx];
    sl[ch][r][cc] = v;
  }
#endif
  __syncthreads();

  // Perceived B fragments (32x16 f16, N=lane&15, K over halves in table order):
  //  frag0: K 0..15 identity, K 16..31 sobel_x; frag1: K 32..47 sobel_y, rest 0.
  v16h bf0, bf1;
  const int cc = 1 + n;
  const int rr = 1 + wv;
  #pragma unroll
  for (int i = 0; i < 16; ++i) {
    const int ch0 = (i & 7) + 8 * hi;
    float idn = sl[ch0][rr][cc];
    float sy = -sl[ch0][rr - 1][cc - 1] - 2.0f * sl[ch0][rr - 1][cc] - sl[ch0][rr - 1][cc + 1]
         +  sl[ch0][rr + 1][cc - 1] + 2.0f * sl[ch0][rr + 1][cc] + sl[ch0][rr + 1][cc + 1];
    float sx = -sl[ch0][rr - 1][cc - 1] + sl[ch0][rr - 1][cc + 1]
         - 2.0f * sl[ch0][rr][cc - 1] + 2.0f * sl[ch0][rr][cc + 1]
         -  sl[ch0][rr + 1][cc - 1] + sl[ch0][rr + 1][cc + 1];
    float v0 = (i < 8) ? idn : sx;
    float v1 = (i < 8) ? sy : 0.0f;
    bf0[i] = (_Float16)v0;
    bf1[i] = (_Float16)v1;
  }

  // Layer 1: h = relu(W1 * perceived + b1); 8 M-tiles x 2 K-chunks
  const v16h* w1f = (const v16h*)w1A;
  #pragma unroll
  for (int mt = 0; mt < 8; ++mt) {
    v8f c;
    #pragma unroll
    for (int r = 0; r < 8; ++r) c[r] = bl1[16 * mt + r + 8 * hi];
    v16h a0 = w1f[(mt * 2 + 0) * 32 + lane];
    v16h a1 = w1f[(mt * 2 + 1) * 32 + lane];
    c = __builtin_amdgcn_wmma_f32_16x16x32_f16(false, a0, false, bf0, (short)0, c, false, false);
    c = __builtin_amdgcn_wmma_f32_16x16x32_f16(false, a1, false, bf1, (short)0, c, false, false);
    #pragma unroll
    for (int r = 0; r < 8; ++r)
      hl[wv][16 * mt + r + 8 * hi][n] = (_Float16)fmaxf(c[r], 0.0f);
  }
  __syncthreads();   // D-layout -> B-layout transpose through LDS

  // Layer 2: dx = W2 * h + b2 ; K=128 in 4 chunks
  const v16h* w2f = (const v16h*)w2A;
  v8f d;
  #pragma unroll
  for (int r = 0; r < 8; ++r) d[r] = bl2[r + 8 * hi];
  #pragma unroll
  for (int j = 0; j < 4; ++j) {
    v16h a  = w2f[j * 32 + lane];
    v16h bh;
    #pragma unroll
    for (int s = 0; s < 16; ++s)
      bh[s] = hl[wv][j * 32 + (s < 8 ? s : s + 8) + 8 * hi][n];
    d = __builtin_amdgcn_wmma_f32_16x16x32_f16(false, a, false, bh, (short)0, d, false, false);
  }

  // residual update with clamp (dropout==0 -> update mask == 1)
  #pragma unroll
  for (int r = 0; r < 8; ++r) {
    const int ch = r + 8 * hi;
    float dx = fminf(fmaxf(d[r], -CLAMP_V), CLAMP_V);
    float sv = sl[ch][1 + wv][1 + n];
    s_mid[((b * C_ + ch) * H_ + y) * W_ + x0 + n] = sv + dx;
  }

  // pre-update alive mask (lanes 0..15 of each wave)
  if (hi == 0) {
    float m = -1e30f;
    #pragma unroll
    for (int dy = 0; dy < 3; ++dy)
      #pragma unroll
      for (int dxx = 0; dxx < 3; ++dxx)
        m = fmaxf(m, sl[3][wv + dy][n + dxx]);
    pre_mask[(b * H_ + y) * W_ + x0 + n] = (m > ALIVE_TH) ? 1 : 0;
  }
}

// ---------------------------------------------------------------------------
// Finalize: post-update alive maxpool, apply (pre & post) mask.
// ---------------------------------------------------------------------------
__global__ void nca_finalize(const float* __restrict__ s_mid,
                             const unsigned char* __restrict__ pre_mask,
                             float* __restrict__ s_out) {
  const int idx = blockIdx.x * blockDim.x + threadIdx.x;   // B*H*W
  const int x = idx & (W_ - 1);
  const int y = (idx >> 7) & (H_ - 1);
  const int b = idx >> 14;

  float m = -1e30f;
  #pragma unroll
  for (int dy = -1; dy <= 1; ++dy) {
    int yy = y + dy;
    if (yy < 0 || yy >= H_) continue;
    #pragma unroll
    for (int dx = -1; dx <= 1; ++dx) {
      int xx = x + dx;
      if (xx < 0 || xx >= W_) continue;
      m = fmaxf(m, s_mid[((b * C_ + 3) * H_ + yy) * W_ + xx]);
    }
  }
  const float keep = (pre_mask[idx] && (m > ALIVE_TH)) ? 1.0f : 0.0f;
  #pragma unroll
  for (int c = 0; c < C_; ++c) {
    float v = s_mid[((b * C_ + c) * H_ + y) * W_ + x];
    s_out[((b * C_ + c) * H_ + y) * W_ + x] = keep * v;
  }
}

// ---------------------------------------------------------------------------
extern "C" void kernel_launch(void* const* d_in, const int* in_sizes, int n_in,
                              void* d_out, int out_size, void* d_ws, size_t ws_size,
                              hipStream_t stream) {
  (void)in_sizes; (void)n_in; (void)out_size; (void)ws_size;
  const float* x  = (const float*)d_in[0];
  const float* w1 = (const float*)d_in[1];
  const float* b1 = (const float*)d_in[2];
  const float* w2 = (const float*)d_in[3];
  const float* b2 = (const float*)d_in[4];
  // num_steps (d_in[5]) is a compile-time-known 8 in the reference.

  char* ws = (char*)d_ws;
  _Float16* w1A       = (_Float16*)(ws);                                  // 16 KB
  _Float16* w2A       = (_Float16*)(ws + (16 << 10));                     // 4 KB
  unsigned char* mask = (unsigned char*)(ws + (20 << 10));                // 128 KB
  float* smid         = (float*)(ws + (20 << 10) + (128 << 10));          // 8 MB
  float* sbuf         = (float*)(ws + (20 << 10) + (128 << 10)
                                 + (size_t)B_ * C_ * H_ * W_ * 4);        // 8 MB

  nca_prep<<<1, 256, 0, stream>>>(w1, w2, w1A, w2A);

  const dim3 grid1(W_ / 16, H_ / 4, B_);
  const float* cur = x;
  for (int step = 0; step < 8; ++step) {
    nca_step<<<grid1, 128, 0, stream>>>(cur, smid, mask, w1A, w2A, b1, b2);
    float* out = (step == 7) ? (float*)d_out : sbuf;
    nca_finalize<<<(B_ * H_ * W_) / 256, 256, 0, stream>>>(smid, mask, out);
    cur = out;
  }
}
